// MultiHeadAttention_62732292325829
// MI455X (gfx1250) — matrix operand order repaired and checked by reading, not compiled, not run on hardware
//
#include <hip/hip_runtime.h>
#include <hip/hip_bf16.h>

#define BB 4
#define SS 1024
#define EE 1024
#define HH 16
#define DD 64
#define NBUCKETS 16
#define NEGINF (-1.0e9f)

typedef __attribute__((ext_vector_type(16))) __bf16 v16bf;
typedef __attribute__((ext_vector_type(8)))  __bf16 v8bf;
typedef __attribute__((ext_vector_type(8)))  float  v8f;
typedef __attribute__((ext_vector_type(4))) unsigned int u32x4;
typedef __attribute__((ext_vector_type(8))) int          i32x8;
typedef __attribute__((ext_vector_type(4))) int          i32x4;

#if defined(__HIP_DEVICE_COMPILE__) && __has_builtin(__builtin_amdgcn_tensor_load_to_lds)
#define USE_TDM 1
#else
#define USE_TDM 0
#endif

static __device__ inline v8f wmma_bf16(v16bf a, v16bf b, v8f c) {
  // D = A(16x32 bf16) * B(32x16 bf16) + C(16x16 f32)
  return __builtin_amdgcn_wmma_f32_16x16x32_bf16(false, a, false, b, (short)0, c, false, false);
}

static __device__ inline v8f zero8() {
  v8f z = {0.f,0.f,0.f,0.f,0.f,0.f,0.f,0.f};
  return z;
}

static __device__ inline void wait_tensorcnt0() {
#if defined(__HIP_DEVICE_COMPILE__)
#if __has_builtin(__builtin_amdgcn_s_wait_tensorcnt)
  __builtin_amdgcn_s_wait_tensorcnt(0);
#else
  asm volatile("s_wait_tensorcnt 0x0" ::: "memory");
#endif
#endif
}

// A-fragment (16x32) from row-major f32 source. p = rowbase + kbase + 8*hi.
// halves 0..7 <- p[0..7], halves 8..15 <- p[16..23]
static __device__ inline v16bf a_frag_from_f32(const float* __restrict__ p) {
  float4 f0 = *(const float4*)(p);
  float4 f1 = *(const float4*)(p + 4);
  float4 f2 = *(const float4*)(p + 16);
  float4 f3 = *(const float4*)(p + 20);
  v16bf a;
  a[0]=(__bf16)f0.x;  a[1]=(__bf16)f0.y;  a[2]=(__bf16)f0.z;  a[3]=(__bf16)f0.w;
  a[4]=(__bf16)f1.x;  a[5]=(__bf16)f1.y;  a[6]=(__bf16)f1.z;  a[7]=(__bf16)f1.w;
  a[8]=(__bf16)f2.x;  a[9]=(__bf16)f2.y;  a[10]=(__bf16)f2.z; a[11]=(__bf16)f2.w;
  a[12]=(__bf16)f3.x; a[13]=(__bf16)f3.y; a[14]=(__bf16)f3.z; a[15]=(__bf16)f3.w;
  return a;
}

// A-fragment (16x32) from row-major bf16 source (global or LDS).
static __device__ inline v16bf a_frag_from_bf16(const __bf16* __restrict__ p) {
  v8bf lo = *(const v8bf*)(p);
  v8bf hi = *(const v8bf*)(p + 16);
  return __builtin_shufflevector(lo, hi, 0,1,2,3,4,5,6,7,8,9,10,11,12,13,14,15);
}

// B-fragment (32x16): lane holds K elements p[0..7], p[16..23] of one column. f32 source.
static __device__ inline v16bf b_frag_from_f32(const float* __restrict__ p) {
  float4 f0 = *(const float4*)(p);
  float4 f1 = *(const float4*)(p + 4);
  float4 f2 = *(const float4*)(p + 16);
  float4 f3 = *(const float4*)(p + 20);
  v16bf b;
  b[0]=(__bf16)f0.x;  b[1]=(__bf16)f0.y;  b[2]=(__bf16)f0.z;  b[3]=(__bf16)f0.w;
  b[4]=(__bf16)f1.x;  b[5]=(__bf16)f1.y;  b[6]=(__bf16)f1.z;  b[7]=(__bf16)f1.w;
  b[8]=(__bf16)f2.x;  b[9]=(__bf16)f2.y;  b[10]=(__bf16)f2.z; b[11]=(__bf16)f2.w;
  b[12]=(__bf16)f3.x; b[13]=(__bf16)f3.y; b[14]=(__bf16)f3.z; b[15]=(__bf16)f3.w;
  return b;
}

// GEMM: out[m,n] = sum_k X[m,k] * W[n,k] + bias[n]    (M=4096, N=K=1024)
// MODE 0: A=f32,  store bf16 row-major (Q, K projections)
// MODE 1: A=f32,  store bf16 transposed per head -> Vt[B,H,D,S]
// MODE 2: A=bf16, store f32 row-major (output projection)
template<int MODE>
__global__ __launch_bounds__(256) void proj_kernel(
    const float* __restrict__ Xf, const __bf16* __restrict__ Xb,
    const float* __restrict__ W, const float* __restrict__ bias,
    __bf16* __restrict__ outb, float* __restrict__ outf)
{
  const int lane = threadIdx.x & 31;
  const int wv   = threadIdx.x >> 5;
  const int hi   = lane >> 4;
  const int ln   = lane & 15;
  const int mbase = blockIdx.x * 128 + wv * 16;   // 8 waves stacked in M
  const int nbase = blockIdx.y * 64;

  v8f c[4];
  c[0] = zero8(); c[1] = zero8(); c[2] = zero8(); c[3] = zero8();

  const int arow = mbase + ln;
  #pragma unroll 2
  for (int kb = 0; kb < EE; kb += 32) {
    v16bf a;
    if (MODE == 2) a = a_frag_from_bf16(Xb + (size_t)arow * EE + kb + 8*hi);
    else           a = a_frag_from_f32 (Xf + (size_t)arow * EE + kb + 8*hi);
    #pragma unroll
    for (int t = 0; t < 4; ++t) {
      const float* wp = W + (size_t)(nbase + t*16 + ln) * EE + kb + 8*hi;
      v16bf bf = b_frag_from_f32(wp);
      c[t] = wmma_bf16(a, bf, c[t]);
    }
  }

  #pragma unroll
  for (int t = 0; t < 4; ++t) {
    const int n = nbase + t*16 + ln;
    const float bv = bias[n];
    #pragma unroll
    for (int r = 0; r < 8; ++r) {
      const int m = mbase + 8*hi + r;
      const float val = c[t][r] + bv;
      if (MODE == 0) {
        outb[(size_t)m * EE + n] = (__bf16)val;
      } else if (MODE == 1) {
        const int bidx = m >> 10, s = m & 1023;
        const int h2 = n >> 6, d = n & 63;
        outb[(((size_t)(bidx*HH + h2))*DD + d)*SS + s] = (__bf16)val;
      } else {
        outf[(size_t)m * EE + n] = val;
      }
    }
  }
}

// One block = (16 q rows, one head, one batch). 8 waves each own a 128-wide k strip.
__global__ __launch_bounds__(256) void attn_kernel(
    const __bf16* __restrict__ Qb, const __bf16* __restrict__ Kb,
    const __bf16* __restrict__ Vt, const float* __restrict__ mask,
    const float* __restrict__ rel_bias,
    float* __restrict__ logits_out, __bf16* __restrict__ ctxb)
{
  __shared__ float  s_logits[16][1032];   // padded f32 logits row block
  __shared__ __bf16 s_P[16][1040];        // padded bf16 un-normalized probs
  __shared__ __bf16 s_Q[16][64];          // Q tile staged by TDM
  __shared__ float  s_ctx[16][64];        // cross-wave ctx accumulator
  __shared__ float  s_red[16][16];
  __shared__ float  s_rowmax[16];
  __shared__ float  s_rowsum[16];

  const int tid  = threadIdx.x;
  const int lane = tid & 31, wv = tid >> 5;
  const int hi   = lane >> 4, ln = lane & 15;
  const int qbase = blockIdx.x * 16;
  const int h = blockIdx.y;
  const int b = blockIdx.z;

  const __bf16* qtile = Qb + ((size_t)(b*SS + qbase))*EE + h*DD;

#if USE_TDM
  if (wv == 0) {
    // Tensor DMA descriptor (CDNA5 ISA):
    // 2D tile: 64 (dim0, bf16) x 16 rows, tensor_dim0_stride = EE.
    const unsigned long long ga = (unsigned long long)(uintptr_t)qtile;
    const unsigned lds_off = (unsigned)(uintptr_t)(&s_Q[0][0]); // flat LDS addr[31:0] = LDS offset
    u32x4 g0;
    g0[0] = 1u;                                   // count=1, is_restore=0, gather off
    g0[1] = lds_off;                              // lds_addr
    g0[2] = (unsigned)(ga & 0xFFFFFFFFull);       // global_addr[31:0]
    g0[3] = (unsigned)((ga >> 32) & 0x1FFFFFFull) // global_addr[56:32]
            | (2u << 30);                         // type=2 (image)
    i32x8 g1;
    g1[0] = (int)(1u << 16);                      // workgroup_mask=0, data_size=1 (2B)
    g1[1] = (int)(64u << 16);                     // tensor_dim0[15:0]=64 (bits 63:48)
    g1[2] = (int)(16u << 16);                     // tensor_dim0[31:16]=0 | tensor_dim1[15:0]=16
    g1[3] = (int)(64u << 16);                     // tensor_dim1[31:16]=0 | tile_dim0=64
    g1[4] = (int)16u;                             // tile_dim1=16, tile_dim2=0
    g1[5] = (int)EE;                              // tensor_dim0_stride[31:0] = 1024
    g1[6] = 0;                                    // stride[47:32]=0, dim1_stride lo=0
    g1[7] = 0;
    i32x4 z4 = {0,0,0,0};
#if defined(__clang_major__) && __clang_major__ >= 23
    i32x8 z8 = {0,0,0,0,0,0,0,0};
    __builtin_amdgcn_tensor_load_to_lds(g0, g1, z4, z4, z8, 0);
#else
    __builtin_amdgcn_tensor_load_to_lds(g0, g1, z4, z4, 0);
#endif
  }
#endif

  for (int i = tid; i < 16*64; i += 256) s_ctx[i >> 6][i & 63] = 0.f;

  wait_tensorcnt0();
  __syncthreads();

  // Q fragments for the two 32-wide d steps (d = 0..63)
#if USE_TDM
  const v16bf aQ0 = a_frag_from_bf16(&s_Q[ln][0  + 8*hi]);
  const v16bf aQ1 = a_frag_from_bf16(&s_Q[ln][32 + 8*hi]);
#else
  const __bf16* qrow = qtile + (size_t)ln * EE;
  const v16bf aQ0 = a_frag_from_bf16(qrow + 0  + 8*hi);
  const v16bf aQ1 = a_frag_from_bf16(qrow + 32 + 8*hi);
#endif

  const int kstrip = wv * 128;
  #pragma unroll
  for (int t = 0; t < 8; ++t) {
    const int kkb = kstrip + t*16;
    const __bf16* kp = Kb + ((size_t)(b*SS + kkb + ln))*EE + h*DD;
    v8f acc = zero8();
    acc = wmma_bf16(aQ0, a_frag_from_bf16(kp + 8*hi),      acc);
    acc = wmma_bf16(aQ1, a_frag_from_bf16(kp + 32 + 8*hi), acc);

    const int kcol = kkb + ln;
    const float mv = mask[b*SS + kcol] * NEGINF;
    #pragma unroll
    for (int r = 0; r < 8; ++r) {
      const int q_idx = qbase + 8*hi + r;
      const int dlt = q_idx - kcol;
      const int ad = dlt < 0 ? -dlt : dlt;
      // branchless T5 bucket: ad<8 ? ad : min(15, 8 + floor(log2(ad/8)*8))
      const float adf = (float)(ad < 8 ? 8 : ad);
      const int lg = 8 + (int)(__log2f(adf * 0.125f) * 8.0f);
      const int lgc = lg < (NBUCKETS - 1) ? lg : (NBUCKETS - 1);
      const int bucket = ad < 8 ? ad : lgc;
      const float pb = rel_bias[bucket * HH + h];
      const float rpm = (bucket == NBUCKETS - 1) ? NEGINF : 0.0f;
      s_logits[8*hi + r][kcol] = (acc[r] + pb) * 0.125f + mv + rpm;
    }
  }
  __syncthreads();

  // Softmax (16 rows x 16 partitions of 64 cols) + coalesced global logits store
  const int row = tid >> 4, part = tid & 15;
  float* grow = logits_out + (((size_t)((b*HH + h)*SS + qbase + row)) * SS) + part*64;
  float mx = -__builtin_inff();
  #pragma unroll 4
  for (int j = 0; j < 64; j += 4) {
    float4 v4 = *(const float4*)&s_logits[row][part*64 + j];
    *(float4*)(grow + j) = v4;
    mx = fmaxf(mx, fmaxf(fmaxf(v4.x, v4.y), fmaxf(v4.z, v4.w)));
  }
  s_red[row][part] = mx;
  __syncthreads();
  if (part == 0) {
    float m2 = -__builtin_inff();
    #pragma unroll
    for (int i = 0; i < 16; ++i) m2 = fmaxf(m2, s_red[row][i]);
    s_rowmax[row] = m2;
  }
  __syncthreads();
  const float rmax = s_rowmax[row];
  float sum = 0.f;
  for (int j = 0; j < 64; ++j) {
    float e = __expf(s_logits[row][part*64 + j] - rmax);
    s_P[row][part*64 + j] = (__bf16)e;
    sum += e;
  }
  s_red[row][part] = sum;
  __syncthreads();
  if (part == 0) {
    float s2 = 0.f;
    #pragma unroll
    for (int i = 0; i < 16; ++i) s2 += s_red[row][i];
    s_rowsum[row] = s2;
  }
  __syncthreads();

  // PV: each wave reduces its 128-wide k strip, accumulate across waves in LDS
  v8f cv[4];
  cv[0] = zero8(); cv[1] = zero8(); cv[2] = zero8(); cv[3] = zero8();
  for (int ks = 0; ks < 128; ks += 32) {
    const int kk = kstrip + ks;
    v16bf aP = a_frag_from_bf16(&s_P[ln][kk + 8*hi]);
    #pragma unroll
    for (int dt = 0; dt < 4; ++dt) {
      const __bf16* vp = Vt + ((size_t)((b*HH + h)*DD + dt*16 + ln))*SS + kk + 8*hi;
      cv[dt] = wmma_bf16(aP, a_frag_from_bf16(vp), cv[dt]);
    }
  }
  #pragma unroll
  for (int dt = 0; dt < 4; ++dt)
    #pragma unroll
    for (int r = 0; r < 8; ++r)
      atomicAdd(&s_ctx[8*hi + r][dt*16 + ln], cv[dt][r]);
  __syncthreads();

  for (int i = tid; i < 16*64; i += 256) {
    const int m = i >> 6, d = i & 63;
    const float v = s_ctx[m][d] / s_rowsum[m];
    ctxb[((size_t)(b*SS + qbase + m))*EE + h*DD + d] = (__bf16)v;
  }
}

extern "C" void kernel_launch(void* const* d_in, const int* in_sizes, int n_in,
                              void* d_out, int out_size, void* d_ws, size_t ws_size,
                              hipStream_t stream) {
  (void)in_sizes; (void)n_in; (void)out_size; (void)ws_size;
  const float* query = (const float*)d_in[0];
  const float* key   = (const float*)d_in[1];
  const float* value = (const float*)d_in[2];
  const float* mask  = (const float*)d_in[3];
  const float* Wq = (const float*)d_in[4];
  const float* bq = (const float*)d_in[5];
  const float* Wk = (const float*)d_in[6];
  const float* bk = (const float*)d_in[7];
  const float* Wv = (const float*)d_in[8];
  const float* bv = (const float*)d_in[9];
  const float* Wo = (const float*)d_in[10];
  const float* bo = (const float*)d_in[11];
  const float* rel_bias = (const float*)d_in[12];

  float* out_f    = (float*)d_out;                 // [B,S,E]
  float* logits_f = out_f + (size_t)BB*SS*EE;      // [B,H,S,S]

  char* ws = (char*)d_ws;                          // 32 MB used
  __bf16* Qb  = (__bf16*)(ws);
  __bf16* Kb  = (__bf16*)(ws + (size_t)BB*SS*EE*2);
  __bf16* Vt  = (__bf16*)(ws + (size_t)BB*SS*EE*4);
  __bf16* Ctx = (__bf16*)(ws + (size_t)BB*SS*EE*6);

  dim3 pb(256);
  dim3 pg(BB*SS/128, EE/64, 1);   // (32,16)
  proj_kernel<0><<<pg, pb, 0, stream>>>(query, nullptr, Wq, bq, Qb, nullptr);
  proj_kernel<0><<<pg, pb, 0, stream>>>(key,   nullptr, Wk, bk, Kb, nullptr);
  proj_kernel<1><<<pg, pb, 0, stream>>>(value, nullptr, Wv, bv, Vt, nullptr);

  dim3 ag(SS/16, HH, BB);         // (64,16,4)
  attn_kernel<<<ag, pb, 0, stream>>>(Qb, Kb, Vt, mask, rel_bias, logits_f, Ctx);

  proj_kernel<2><<<pg, pb, 0, stream>>>(nullptr, Ctx, Wo, bo, nullptr, out_f);
}
